// SelfAttention_86517821213816
// MI455X (gfx1250) — matrix operand order repaired and checked by reading, not compiled, not run on hardware
//
#include <hip/hip_runtime.h>

// B=8, D=128, L=2048, H=8, DH=16
#define Bc 8
#define Dc 128
#define Lc 2048
#define Hc 8

typedef __attribute__((ext_vector_type(16))) __bf16 v16bf;
typedef __attribute__((ext_vector_type(8)))  __bf16 v8bf;
typedef __attribute__((ext_vector_type(8)))  float  v8f;
typedef __attribute__((ext_vector_type(4)))  unsigned int v4u;
typedef __attribute__((ext_vector_type(8)))  int v8i;
typedef __attribute__((ext_vector_type(4)))  int v4i;

#if defined(__has_builtin)
#if __has_builtin(__builtin_amdgcn_tensor_load_to_lds) && \
    __has_builtin(__builtin_amdgcn_s_wait_tensorcnt)
#define USE_TDM 1
#endif
#endif
#ifndef USE_TDM
#define USE_TDM 0
#endif

static __device__ __forceinline__ v8f wmma_bf16(v16bf a, v16bf b, v8f c) {
  // D = A(16x32 bf16) x B(32x16 bf16) + C(16x16 f32)
  return __builtin_amdgcn_wmma_f32_16x16x32_bf16(false, a, false, b, (short)0, c,
                                                 false, false);
}

#if USE_TDM
// 2D TDM load: copies a [dim1 x dim0] (rows x cols, data_size=2B) tile whose
// tile dims == tensor dims (flat copy) from global to LDS. D# per ISA 08 §8.
static __device__ __forceinline__ void tdm_load_2d(const void* gptr,
                                                   unsigned lds_off,
                                                   unsigned dim0, unsigned dim1,
                                                   unsigned stride0) {
  const unsigned long long ga = (unsigned long long)gptr;
  v4u g0;
  g0[0] = 1u;                                    // count=1, no gather/restore
  g0[1] = lds_off;                               // lds_addr (bytes)
  g0[2] = (unsigned)(ga & 0xFFFFFFFFu);          // global_addr[31:0]
  g0[3] = (unsigned)((ga >> 32) & 0x01FFFFFFu)   // global_addr[56:32]
          | 0x80000000u;                         // type=2 ("image")
  v8i g1;
  g1[0] = (int)(1u << 16);                       // wg_mask=0, data_size=1 (2B)
  g1[1] = (int)(dim0 << 16);                     // tensor_dim0[15:0] @ bit48
  g1[2] = (int)((dim0 >> 16) | (dim1 << 16));    // dim0 hi | tensor_dim1 lo
  g1[3] = (int)((dim1 >> 16) | (dim0 << 16));    // dim1 hi | tile_dim0 = dim0
  g1[4] = (int)dim1;                             // tile_dim1 = dim1, tile_dim2=0
  g1[5] = (int)stride0;                          // tensor_dim0_stride[31:0]
  g1[6] = 0;                                     // stride0 hi | stride1 lo
  g1[7] = 0;
  v4i gz = {0, 0, 0, 0};                         // groups 2/3 unused (2D)
#if defined(__clang_major__) && (__clang_major__ >= 23)
  v8i gz8 = {0, 0, 0, 0, 0, 0, 0, 0};
  __builtin_amdgcn_tensor_load_to_lds(g0, g1, gz, gz, gz8, 0);
#else
  __builtin_amdgcn_tensor_load_to_lds(g0, g1, gz, gz, 0);
#endif
}
#endif

// ---------------------------------------------------------------------------
// Kernel 1: fused projection GEMM  [B*L x 128] @ [128 x 384]  ->  bf16 K/V/Q
//   channels   0..127 : K   -> Km[b][h][l][dh]      (row-major per head)
//   channels 128..255 : V   -> Vt[b][h][dh][l]      (transposed: WMMA-B layout)
//   channels 256..383 : Q   -> Qs[b][h][l][dh] * DH^-0.5
// One wave32 per 16x16 output tile, K=128 as 4 chained 16x16x32 bf16 WMMAs.
// ---------------------------------------------------------------------------
__global__ __launch_bounds__(32) void proj_wmma(
    const float* __restrict__ qin,   // queries [B][D][L] (channels-first)
    const float* __restrict__ wmem,  // [2D][D]
    const float* __restrict__ wq,    // [D][D]
    __bf16* __restrict__ Qs, __bf16* __restrict__ Km, __bf16* __restrict__ Vt) {
  const int tile  = blockIdx.x;
  const int ntile = tile % 24;            // 16-channel tile of the 384 outputs
  const int mtile = tile / 24;            // 16-row tile over b*L
  const int b     = mtile >> 7;           // / (2048/16)
  const int l0    = (mtile & 127) << 4;
  const int lane  = threadIdx.x & 31;
  const int half  = lane >> 4;
  const int lm    = lane & 15;

  const int e = ntile * 16 + lm;          // this lane's output channel (B lane = N)
  const float* wrow = (e < 2 * Dc) ? (wmem + (size_t)e * Dc)
                                   : (wq + (size_t)(e - 2 * Dc) * Dc);
  // x[b, l0+lm, d] = queries[b*D*L + d*L + (l0+lm)]  (A lane = M row)
  const float* xcol = qin + (size_t)b * Dc * Lc + (l0 + lm);

  v8f c = {};
#pragma unroll
  for (int k0 = 0; k0 < Dc; k0 += 32) {
    v16bf a, bb;
    // A lane layout: elems 0..7 -> K = k0+8h..+7 ; elems 8..15 -> K = k0+16+8h..
#pragma unroll
    for (int i = 0; i < 8; ++i) {
      a[i]     = (__bf16)xcol[(size_t)(k0 + half * 8 + i) * Lc];
      a[i + 8] = (__bf16)xcol[(size_t)(k0 + 16 + half * 8 + i) * Lc];
    }
    // B lane layout: elems 0..7 -> K = k0+8h..+7 ; elems 8..15 -> K = k0+16+8h..
#pragma unroll
    for (int i = 0; i < 16; ++i) bb[i] = (__bf16)wrow[k0 + (i < 8 ? half * 8 + i : 16 + half * 8 + (i - 8))];
    c = wmma_bf16(a, bb, c);
  }

  // D tile: VGPR r, row M = r + 8*half, col N = lm (channel e)
  if (ntile < 8) {                               // K, head = ntile
    __bf16* kb = Km + ((size_t)b * Hc + ntile) * Lc * 16;
#pragma unroll
    for (int r = 0; r < 8; ++r)
      kb[(size_t)(l0 + r + 8 * half) * 16 + lm] = (__bf16)c[r];
  } else if (ntile < 16) {                       // V, head = ntile-8, transposed
    __bf16* vb = Vt + (((size_t)b * Hc + (ntile - 8)) * 16 + lm) * Lc;
#pragma unroll
    for (int r = 0; r < 8; ++r) vb[l0 + r + 8 * half] = (__bf16)c[r];
  } else {                                       // Q, head = ntile-16, pre-scaled
    __bf16* qb = Qs + ((size_t)b * Hc + (ntile - 16)) * Lc * 16;
#pragma unroll
    for (int r = 0; r < 8; ++r)
      qb[(size_t)(l0 + r + 8 * half) * 16 + lm] = (__bf16)(0.25f * c[r]);
  }
}

// ---------------------------------------------------------------------------
// Kernel 2: attention. 8-wave workgroup per (b, h, group of 8 query tiles).
// K block [2048][16] (64KB) + V block [16][2048] (64KB) are bulk-staged into
// LDS (TDM tensor_load_to_lds when available, cooperative copy otherwise) and
// shared by all 8 waves. Softmax is unnormalized-exp (shift-invariant; logits
// are O(10) here) so the only cross-lane reduction is one row-sum at the end.
// Per 32-key chunk: 2 logit WMMAs -> mask/exp -> P via LDS -> 1 P@V WMMA.
// ---------------------------------------------------------------------------
__global__ __launch_bounds__(256) void attn_wmma(
    const __bf16* __restrict__ Qs, const __bf16* __restrict__ Km,
    const __bf16* __restrict__ Vt, const int* __restrict__ mask,
    float* __restrict__ out) {
  extern __shared__ __align__(16) char smem[];
  __bf16* Kl = (__bf16*)smem;              // [2048][16] bf16, 64KB
  __bf16* Vl = (__bf16*)(smem + 65536);    // [16][2048] bf16, 64KB

  const int g    = blockIdx.x;             // 0..15 : group of 8 q-tiles
  const int bh   = blockIdx.y;             // 0..63 : b*H + h
  const int h    = bh & 7;
  const int b    = bh >> 3;
  const int tid  = threadIdx.x;
  const int wave = tid >> 5;
  const int lane = tid & 31;
  const int half = lane >> 4;
  const int lm   = lane & 15;
  const int l0   = (g * 8 + wave) * 16;    // this wave's 16 query rows

  __bf16* Pl = (__bf16*)(smem + 131072) + (size_t)wave * 512;  // [16][32] per wave

  const __bf16* Qbh = Qs + (size_t)bh * Lc * 16;
  const __bf16* Kbh = Km + (size_t)bh * Lc * 16;
  const __bf16* Vbh = Vt + (size_t)bh * 16 * Lc;
  const int* mrow   = mask + (size_t)b * Lc;

  // ---- stage K and V blocks into LDS ----
#if USE_TDM
  if (wave == 0) {
    tdm_load_2d(Kbh, (unsigned)(unsigned long long)(void*)Kl, 16u, 2048u, 16u);
    tdm_load_2d(Vbh, (unsigned)(unsigned long long)(void*)Vl, 2048u, 16u, 2048u);
    __builtin_amdgcn_s_wait_tensorcnt(0);
  }
#else
  {
    const uint4* sK = (const uint4*)Kbh;
    const uint4* sV = (const uint4*)Vbh;
    uint4* dK = (uint4*)Kl;
    uint4* dV = (uint4*)Vl;
    for (int i = tid; i < 4096; i += 256) {  // 4096 * 16B = 64KB each
      dK[i] = sK[i];
      dV[i] = sV[i];
    }
  }
#endif
  __syncthreads();

  // ---- persistent Q A-operand: lane m holds dh 8h..8h+7; padded K-half zero
  v16bf qa;
#pragma unroll
  for (int i = 0; i < 16; ++i) qa[i] = (__bf16)0.0f;
  {
    const v8bf qv = *(const v8bf*)(Qbh + (size_t)(l0 + lm) * 16 + half * 8);
#pragma unroll
    for (int i = 0; i < 8; ++i) qa[i] = qv[i];
  }

  v8f acc = {};
  float psum[8];
#pragma unroll
  for (int r = 0; r < 8; ++r) psum[r] = 0.0f;

  for (int kc = 0; kc < Lc; kc += 32) {
#pragma unroll
    for (int t = 0; t < 2; ++t) {
      const int key = kc + t * 16 + lm;
      // B = K^T: lane half h holds Kl[key][8h..8h+7] in elems 0..7 (K rows
      // 8h..8h+7); elems 8..15 carry the zero padded K rows 16..31.
      v16bf kb;
#pragma unroll
      for (int i = 0; i < 16; ++i) kb[i] = (__bf16)0.0f;
      {
        const v8bf kv = *(const v8bf*)(Kl + (size_t)key * 16 + half * 8);
#pragma unroll
        for (int i = 0; i < 8; ++i) kb[i] = kv[i];
      }
      v8f z = {};
      z = wmma_bf16(qa, kb, z);
      const bool keep = (mrow[key] != 0);  // logits*m + (1-m)*(-1e30) -> exp=0
#pragma unroll
      for (int r = 0; r < 8; ++r) {
        const float p = keep ? __expf(z[r]) : 0.0f;
        psum[r] += p;                       // lane owns fixed key columns
        Pl[(r + 8 * half) * 32 + t * 16 + lm] = (__bf16)p;
      }
    }

    // same-wave LDS RAW: DS is in-order per wave; wait as cheap insurance
    asm volatile("s_wait_dscnt 0" ::: "memory");

    // re-load P in A layout: lane m, K halves {8h..8h+7, 16+8h..16+8h+7}
    union { v16bf v; v8bf hh[2]; } pa;
    pa.hh[0] = *(const v8bf*)(Pl + lm * 32 + half * 8);
    pa.hh[1] = *(const v8bf*)(Pl + lm * 32 + 16 + half * 8);
    // V B-operand: lane n=dh holds Vl[dh][kc+8h..+7] and Vl[dh][kc+16+8h..+7]
    union { v16bf v; v8bf hh[2]; } vbv;
    vbv.hh[0] = *(const v8bf*)(Vl + (size_t)lm * Lc + kc + half * 8);
    vbv.hh[1] = *(const v8bf*)(Vl + (size_t)lm * Lc + kc + 16 + half * 8);
    acc = wmma_bf16(pa.v, vbv.v, acc);
    asm volatile("" ::: "memory");
  }

  // one final row-sum reduction across the 16 lanes of each half
#pragma unroll
  for (int m = 8; m >= 1; m >>= 1)
#pragma unroll
    for (int r = 0; r < 8; ++r) psum[r] += __shfl_xor(psum[r], m, 32);

  // out[b][h*16+dh][l0 + r + 8*half] = acc / rowsum   (output is [B, D, L])
  float* ob = out + ((size_t)b * Dc + h * 16 + lm) * Lc + l0 + 8 * half;
#pragma unroll
  for (int r = 0; r < 8; ++r) ob[r] = acc[r] / psum[r];
}

extern "C" void kernel_launch(void* const* d_in, const int* in_sizes, int n_in,
                              void* d_out, int out_size, void* d_ws, size_t ws_size,
                              hipStream_t stream) {
  const float* queries = (const float*)d_in[0];  // [B, D, L] f32
  const int*   mask    = (const int*)d_in[1];    // [B, L] i32
  const float* W_mem   = (const float*)d_in[2];  // [2D, D] f32
  const float* W_q     = (const float*)d_in[3];  // [D, D] f32
  float*       out     = (float*)d_out;          // [B, D, L] f32

  // Workspace: bf16 Qs | Km | Vt, 4 MB each (12 MB total)
  const size_t per = (size_t)Bc * Hc * Lc * 16;  // elements per tensor
  __bf16* Qs = (__bf16*)d_ws;
  __bf16* Km = Qs + per;
  __bf16* Vt = Km + per;

  // 1) projections: (B*L/16) * (384/16) = 1024*24 wave32 tiles
  proj_wmma<<<1024 * 24, 32, 0, stream>>>(queries, W_mem, W_q, Qs, Km, Vt);

  // 2) attention: 64 (b,h) x 16 groups, 8 waves each; 136KB dynamic LDS
  const size_t shmem = 65536 + 65536 + 8 * 1024;
  attn_wmma<<<dim3(16, 64), 256, shmem, stream>>>(Qs, Km, Vt, mask, out);
}
